// AttentionTemporalEncoder_5119601016838
// MI455X (gfx1250) — matrix operand order repaired and checked by reading, not compiled, not run on hardware
//
#include <hip/hip_runtime.h>

// ---------------------------------------------------------------------------
// AttentionTemporalEncoder for MI455X (gfx1250, wave32, WMMA).
// GEMMs via v_wmma_f32_16x16x32_f16; attention K/V tiles double-buffered in
// LDS via GLOBAL_LOAD_ASYNC_TO_LDS_B128 (ASYNCcnt-tracked async copies).
// ---------------------------------------------------------------------------

#define NH   16
#define HD   64
#define HID  1024
#define BSZ  32
#define TSEQ 512

typedef __attribute__((ext_vector_type(16))) _Float16 v16h;
typedef __attribute__((ext_vector_type(8)))  _Float16 v8h;
typedef __attribute__((ext_vector_type(8)))  float    v8f;

union V16U { v16h v; v8h h[2]; };

// Build a 16-half fragment from two 16-byte-aligned 8-half chunks.
__device__ __forceinline__ v16h load_frag(const _Float16* p0, const _Float16* p1) {
    V16U u;
    u.h[0] = *(const v8h*)p0;
    u.h[1] = *(const v8h*)p1;
    return u.v;
}

__device__ __forceinline__ v8f wmma16(v16h a, v16h b, v8f c) {
    // D = A(16x32 f16) * B(32x16 f16) + C(16x16 f32)
    return __builtin_amdgcn_wmma_f32_16x16x32_f16(
        /*neg_a=*/false, a, /*neg_b=*/false, b,
        /*c_mod=*/(short)0, c, /*reuse_a=*/false, /*reuse_b=*/false);
}

// Async 16B copy global -> LDS, tracked by ASYNCcnt (per-lane addresses).
__device__ __forceinline__ void async_copy_b128(const _Float16* gp, unsigned lds_addr) {
    asm volatile("global_load_async_to_lds_b128 %0, %1, off"
                 :: "v"(lds_addr), "v"(gp) : "memory");
}

// ---------------------------------------------------------------------------
// Kernel 0a: x[B, HID, T] f32  ->  xt[B, T, HID] f16 (LDS-tiled transpose)
// ---------------------------------------------------------------------------
__global__ __launch_bounds__(1024) void xpose_kernel(const float* __restrict__ x,
                                                     _Float16* __restrict__ xt) {
    __shared__ float tile[32][33];
    const int b  = blockIdx.z;
    const int t0 = blockIdx.x * 32;
    const int h0 = blockIdx.y * 32;
    const int tx = threadIdx.x, ty = threadIdx.y;
    tile[ty][tx] = x[((long)b * HID + (h0 + ty)) * TSEQ + (t0 + tx)];
    __syncthreads();
    xt[((long)b * TSEQ + (t0 + ty)) * HID + (h0 + tx)] = (_Float16)tile[tx][ty];
}

// ---------------------------------------------------------------------------
// Kernel 0b: f32 -> f16 elementwise (weights)
// ---------------------------------------------------------------------------
__global__ __launch_bounds__(256) void cvt_kernel(const float* __restrict__ src,
                                                  _Float16* __restrict__ dst, int n) {
    int i = blockIdx.x * blockDim.x + threadIdx.x;
    if (i < n) dst[i] = (_Float16)src[i];
}

// ---------------------------------------------------------------------------
// Kernel 1: QKV projection.  y = xt @ W^T + bias.
// Q,K stored f16 [B*T, HID]; V stored TRANSPOSED as vT[b][head][d][t] so the
// attention kernel can async-copy V tiles row-wise into LDS ([d][k] layout).
// Grid: (B*T/128, HID/64, 3 {q,k,v}), block 128. Each wave owns TWO 16-row
// strips so every B-fragment (weights) is reused for 2 WMMAs.
// ---------------------------------------------------------------------------
__global__ __launch_bounds__(128) void qkv_kernel(
    const _Float16* __restrict__ xt,
    const _Float16* __restrict__ Wq, const _Float16* __restrict__ Wk,
    const _Float16* __restrict__ Wv,
    const float* __restrict__ bq, const float* __restrict__ bk,
    const float* __restrict__ bv,
    _Float16* __restrict__ qo, _Float16* __restrict__ ko,
    _Float16* __restrict__ vo) {
    const _Float16* W; const float* bias; _Float16* out;
    if (blockIdx.z == 0)      { W = Wq; bias = bq; out = qo; }
    else if (blockIdx.z == 1) { W = Wk; bias = bk; out = ko; }
    else                      { W = Wv; bias = bv; out = vo; }
    const bool vtr = (blockIdx.z == 2);

    const int lane  = threadIdx.x & 31;
    const int wave  = threadIdx.x >> 5;
    const int m     = lane & 15;
    const int kbA   = (lane & 16) ? 8 : 0;
    const int kbB   = (lane & 16) ? 8 : 0;
    const int rowhi = (lane & 16) ? 8 : 0;
    const long mt   = blockIdx.x;                // 128-token tile over B*T
    const int  o0   = blockIdx.y * 64;

    const _Float16* arow0 = xt + (mt * 128 + wave * 32 + m) * HID;
    const _Float16* arow1 = arow0 + 16 * HID;

    v8f acc[2][4] = {};
    for (int kk = 0; kk < HID; kk += 32) {
        const v16h a0 = load_frag(arow0 + kk + kbA, arow0 + kk + 16 + kbA);
        const v16h a1 = load_frag(arow1 + kk + kbA, arow1 + kk + 16 + kbA);
        __builtin_prefetch(arow0 + kk + 128, 0, 1);   // global_prefetch_b8
#pragma unroll
        for (int n = 0; n < 4; ++n) {
            const _Float16* brow = W + (long)(o0 + 16 * n + m) * HID + kk + kbB;
            const v16h bf = load_frag(brow, brow + 16);
            acc[0][n] = wmma16(a0, bf, acc[0][n]);
            acc[1][n] = wmma16(a1, bf, acc[1][n]);
        }
    }
    float bval[4];
#pragma unroll
    for (int n = 0; n < 4; ++n) bval[n] = bias[o0 + 16 * n + m];
#pragma unroll
    for (int ss = 0; ss < 2; ++ss)
#pragma unroll
        for (int n = 0; n < 4; ++n)
#pragma unroll
            for (int r = 0; r < 8; ++r) {
                const long tok = mt * 128 + wave * 32 + ss * 16 + r + rowhi;
                const _Float16 hv = (_Float16)(acc[ss][n][r] + bval[n]);
                if (!vtr) {
                    out[tok * HID + o0 + 16 * n + m] = hv;
                } else {
                    const int o  = o0 + 16 * n + m;
                    const int bb = (int)(tok >> 9);         // /TSEQ
                    const int tl = (int)(tok & (TSEQ - 1));
                    out[((long)(bb * NH + (o >> 6)) * HD + (o & 63)) * TSEQ + tl] = hv;
                }
            }
}

// ---------------------------------------------------------------------------
// Kernel 2: flash attention per (b, head, 64-query tile).
// Block 128 (4 waves, 16 query rows each). K tile [k][d] and V tile [d][k]
// double-buffered in LDS via async global->LDS b128 copies; tile kt+1 is
// prefetched while tile kt is consumed. Each wave issues 8 async instructions
// per tile pair, so s_wait_asynccnt 8 gates exactly on the current tile.
// ---------------------------------------------------------------------------
__global__ __launch_bounds__(128) void attn_kernel(
    const _Float16* __restrict__ qb, const _Float16* __restrict__ kbuf,
    const _Float16* __restrict__ vt, const int* __restrict__ mask,
    _Float16* __restrict__ attn_out) {
    __shared__ __align__(16) _Float16 Ktile[2][64][72];   // [buf][k_tok][d]
    __shared__ __align__(16) _Float16 Vtile[2][64][72];   // [buf][d][k_tok]
    __shared__ __align__(16) _Float16 Plds[4][16][72];    // per-wave P round-trip

    const int b    = blockIdx.z;
    const int head = blockIdx.y;
    const int qt   = blockIdx.x;
    const int lane  = threadIdx.x & 31;
    const int wave  = threadIdx.x >> 5;
    const int m     = lane & 15;
    const int kbA   = (lane & 16) ? 8 : 0;
    const int kbB   = (lane & 16) ? 8 : 0;
    const int rowhi = (lane & 16) ? 8 : 0;
    const float scale = 0.125f;                  // 1/sqrt(HD)

    const long t0 = (long)b * TSEQ + qt * 64;

    // Issue one K+V tile's async copies (4 K + 4 V instructions per wave).
    auto issue_tile = [&](int kt_, int buf_) {
        const long tk0 = (long)b * TSEQ + (long)kt_ * 64;
        const _Float16* ksrc = kbuf + tk0 * HID + head * HD;                 // row: +HID
        const _Float16* vsrc = vt + ((long)(b * NH + head) * HD) * TSEQ + (long)kt_ * 64; // row: +TSEQ
#pragma unroll
        for (int c = 0; c < 4; ++c) {
            const int ch  = (int)threadIdx.x + 128 * c;   // 0..511 chunk id
            const int row = ch >> 3;
            const int cb  = (ch & 7) * 8;                 // halves
            async_copy_b128(ksrc + (long)row * HID + cb,
                            (unsigned)(unsigned long long)&Ktile[buf_][row][cb]);
            async_copy_b128(vsrc + (long)row * TSEQ + cb,
                            (unsigned)(unsigned long long)&Vtile[buf_][row][cb]);
        }
    };

    // Resident Q fragments: d = 0..31 and d = 32..63.
    const _Float16* qrow = qb + (t0 + wave * 16 + m) * HID + head * HD;
    const v16h qa0 = load_frag(qrow + 0  + kbA, qrow + 16 + kbA);
    const v16h qa1 = load_frag(qrow + 32 + kbA, qrow + 48 + kbA);

    float m_i[8], l_i[8];
#pragma unroll
    for (int r = 0; r < 8; ++r) { m_i[r] = -1e30f; l_i[r] = 0.f; }
    v8f o_acc[4] = {};

    issue_tile(0, 0);                            // prologue prefetch

    for (int kt = 0; kt < TSEQ / 64; ++kt) {
        const int buf = kt & 1;
        if (kt + 1 < TSEQ / 64) {
            issue_tile(kt + 1, buf ^ 1);
            asm volatile("s_wait_asynccnt 0x8" ::: "memory");  // current tile landed
        } else {
            asm volatile("s_wait_asynccnt 0x0" ::: "memory");
        }
        __syncthreads();                          // all waves' chunks visible

        int mk[4];
#pragma unroll
        for (int n = 0; n < 4; ++n) mk[n] = mask[b * TSEQ + kt * 64 + 16 * n + m];

        // --- scores S = Q x K^T (16 x 64), K from LDS as B-fragments ---
        v8f s[4] = {};
#pragma unroll
        for (int n = 0; n < 4; ++n) {
            const _Float16* kr = &Ktile[buf][16 * n + m][0];
            s[n] = wmma16(qa0, load_frag(kr + kbB, kr + kbB + 16), s[n]);
            s[n] = wmma16(qa1, load_frag(kr + 32 + kbB, kr + 32 + kbB + 16), s[n]);
        }
        // scale, clamp to [-50,50], mask==0 -> -50  (matches reference exactly)
#pragma unroll
        for (int n = 0; n < 4; ++n)
#pragma unroll
            for (int r = 0; r < 8; ++r) {
                float val = fminf(fmaxf(s[n][r] * scale, -50.f), 50.f);
                if (mk[n] == 0) val = -50.f;
                s[n][r] = val;
            }
        // --- online softmax update (rows live in 16-lane halves) ---
#pragma unroll
        for (int r = 0; r < 8; ++r) {
            float rm = fmaxf(fmaxf(s[0][r], s[1][r]), fmaxf(s[2][r], s[3][r]));
#pragma unroll
            for (int off = 1; off < 16; off <<= 1)
                rm = fmaxf(rm, __shfl_xor(rm, off, 32));
            const float mnew = fmaxf(m_i[r], rm);
            const float corr = __expf(m_i[r] - mnew);
            m_i[r] = mnew;
            float rs = 0.f;
#pragma unroll
            for (int n = 0; n < 4; ++n) {
                const float p = __expf(s[n][r] - mnew);
                s[n][r] = p;
                rs += p;
            }
#pragma unroll
            for (int off = 1; off < 16; off <<= 1)
                rs += __shfl_xor(rs, off, 32);
            l_i[r] = l_i[r] * corr + rs;
#pragma unroll
            for (int n = 0; n < 4; ++n) o_acc[n][r] *= corr;
        }
        // --- P (D-layout) -> LDS f16 -> reload as A-fragments ---
#pragma unroll
        for (int n = 0; n < 4; ++n)
#pragma unroll
            for (int r = 0; r < 8; ++r)
                Plds[wave][r + rowhi][16 * n + m] = (_Float16)s[n][r];

        const _Float16* pr = &Plds[wave][m][0];
        const v16h pa0 = load_frag(pr + 0  + kbA, pr + 16 + kbA);
        const v16h pa1 = load_frag(pr + 32 + kbA, pr + 48 + kbA);

        // --- O += P x V, V tile already [d][k] in LDS -> B-fragments ---
#pragma unroll
        for (int nd = 0; nd < 4; ++nd) {
            const _Float16* vr = &Vtile[buf][16 * nd + m][0];
            o_acc[nd] = wmma16(pa0, load_frag(vr + kbB, vr + kbB + 16), o_acc[nd]);
            o_acc[nd] = wmma16(pa1, load_frag(vr + 32 + kbB, vr + 32 + kbB + 16), o_acc[nd]);
        }
        __syncthreads();   // everyone done with buf before it is re-prefetched
    }
    // normalize and store [B*T, HID] f16
#pragma unroll
    for (int nd = 0; nd < 4; ++nd)
#pragma unroll
        for (int r = 0; r < 8; ++r) {
            const float ov = o_acc[nd][r] / l_i[r];
            const long tok = t0 + wave * 16 + r + rowhi;
            attn_out[tok * HID + head * HD + 16 * nd + m] = (_Float16)ov;
        }
}

// ---------------------------------------------------------------------------
// Kernel 3: out[b, o] = max_t ( attn[b,t,:] @ Wo[o,:] + bo[o] )
// Grid: (HID/64, B), block 256 (8 waves). Each wave processes strip PAIRS so
// every Wo B-fragment feeds 2 WMMAs; per-lane column max + shfl + LDS reduce.
// ---------------------------------------------------------------------------
__global__ __launch_bounds__(256) void proj_max_kernel(
    const _Float16* __restrict__ attn, const _Float16* __restrict__ Wo,
    const float* __restrict__ bo, float* __restrict__ out) {
    __shared__ float red[8][64];
    const int o0 = blockIdx.x * 64;
    const int b  = blockIdx.y;
    const int lane = threadIdx.x & 31;
    const int wave = threadIdx.x >> 5;
    const int m    = lane & 15;
    const int kbA  = (lane & 16) ? 8 : 0;
    const int kbB  = (lane & 16) ? 8 : 0;

    float cmax[4] = {-1e30f, -1e30f, -1e30f, -1e30f};

#pragma unroll
    for (int p = 0; p < 2; ++p) {
        const int pi = wave + p * 8;              // pair 0..15 -> strips 2pi, 2pi+1
        const _Float16* arow0 = attn + ((long)b * TSEQ + pi * 32 + m) * HID;
        const _Float16* arow1 = arow0 + 16 * HID;
        v8f acc[2][4] = {};
        for (int kk = 0; kk < HID; kk += 32) {
            const v16h a0 = load_frag(arow0 + kk + kbA, arow0 + kk + 16 + kbA);
            const v16h a1 = load_frag(arow1 + kk + kbA, arow1 + kk + 16 + kbA);
#pragma unroll
            for (int n = 0; n < 4; ++n) {
                const _Float16* brow = Wo + (long)(o0 + 16 * n + m) * HID + kk + kbB;
                const v16h bf = load_frag(brow, brow + 16);
                acc[0][n] = wmma16(a0, bf, acc[0][n]);
                acc[1][n] = wmma16(a1, bf, acc[1][n]);
            }
        }
#pragma unroll
        for (int n = 0; n < 4; ++n) {
            float v = fmaxf(acc[0][n][0], acc[1][n][0]);
#pragma unroll
            for (int r = 1; r < 8; ++r)
                v = fmaxf(v, fmaxf(acc[0][n][r], acc[1][n][r]));
            v = fmaxf(v, __shfl_xor(v, 16, 32));  // merge rows r and r+8 groups
            cmax[n] = fmaxf(cmax[n], v);
        }
    }
    if (lane < 16)
#pragma unroll
        for (int n = 0; n < 4; ++n)
            red[wave][16 * n + lane] = cmax[n] + bo[o0 + 16 * n + lane];
    __syncthreads();
    if (threadIdx.x < 64) {
        float v = red[0][threadIdx.x];
#pragma unroll
        for (int w = 1; w < 8; ++w) v = fmaxf(v, red[w][threadIdx.x]);
        out[(long)b * HID + o0 + threadIdx.x] = v;
    }
}

// ---------------------------------------------------------------------------
// Host launcher. Workspace layout (f16):
//   [0,      S)  xt  (reused as attn_out after QKV consumes it)
//   [S,     2S)  q      [2S, 3S) k      [3S, 4S) vT[b][head][d][t]
//   [4S, 4S+8MB) Wq16, Wk16, Wv16, Wo16 (2MB each)
//   S = B*T*HID*2 = 32 MB; total = 136 MB.
// ---------------------------------------------------------------------------
extern "C" void kernel_launch(void* const* d_in, const int* in_sizes, int n_in,
                              void* d_out, int out_size, void* d_ws, size_t ws_size,
                              hipStream_t stream) {
    (void)in_sizes; (void)n_in; (void)out_size; (void)ws_size;
    const float* x    = (const float*)d_in[0];
    const int*   mask = (const int*)d_in[1];
    const float* Wq   = (const float*)d_in[2];
    const float* bq   = (const float*)d_in[3];
    const float* Wk   = (const float*)d_in[4];
    const float* bk   = (const float*)d_in[5];
    const float* Wv   = (const float*)d_in[6];
    const float* bv   = (const float*)d_in[7];
    const float* Wo   = (const float*)d_in[8];
    const float* bo   = (const float*)d_in[9];
    float* out = (float*)d_out;

    char* ws = (char*)d_ws;
    const size_t S = (size_t)BSZ * TSEQ * HID * sizeof(_Float16);
    _Float16* xt   = (_Float16*)(ws);
    _Float16* qf   = (_Float16*)(ws + S);
    _Float16* kf   = (_Float16*)(ws + 2 * S);
    _Float16* vf   = (_Float16*)(ws + 3 * S);   // transposed V
    _Float16* wq16 = (_Float16*)(ws + 4 * S);
    _Float16* wk16 = wq16 + (size_t)HID * HID;
    _Float16* wv16 = wk16 + (size_t)HID * HID;
    _Float16* wo16 = wv16 + (size_t)HID * HID;

    // 0a: transpose + f16 convert x
    xpose_kernel<<<dim3(TSEQ / 32, HID / 32, BSZ), dim3(32, 32, 1), 0, stream>>>(x, xt);
    // 0b: weight conversions
    const int nw = HID * HID;
    cvt_kernel<<<(nw + 255) / 256, 256, 0, stream>>>(Wq, wq16, nw);
    cvt_kernel<<<(nw + 255) / 256, 256, 0, stream>>>(Wk, wk16, nw);
    cvt_kernel<<<(nw + 255) / 256, 256, 0, stream>>>(Wv, wv16, nw);
    cvt_kernel<<<(nw + 255) / 256, 256, 0, stream>>>(Wo, wo16, nw);
    // 1: QKV projection (WMMA, 2x B-fragment reuse; V stored transposed)
    qkv_kernel<<<dim3(BSZ * TSEQ / 128, HID / 64, 3), 128, 0, stream>>>(
        xt, wq16, wk16, wv16, bq, bk, bv, qf, kf, vf);
    // 2: flash attention (WMMA + async double-buffered LDS tiles)
    attn_kernel<<<dim3(TSEQ / 64, NH, BSZ), 128, 0, stream>>>(qf, kf, vf, mask, xt);
    // 3: output projection + max over T (WMMA, 2x B-fragment reuse)
    proj_max_kernel<<<dim3(HID / 64, BSZ), 256, 0, stream>>>(xt, wo16, bo, out);
}
